// GraphTransformer_29618094473491
// MI455X (gfx1250) — matrix operand order repaired and checked by reading, not compiled, not run on hardware
//
#include <hip/hip_runtime.h>
#include <math.h>

// GraphTransformer: 2x TransformerConv (heads=4,ch=32 then heads=1,ch=2)
// Layer-1 q/k/v/skip GEMMs via v_wmma_f32_16x16x32_f16 (f32 accumulate).
// Edge attention via L2-resident gathers + encoded-uint atomicMax segment-max
// + f32 atomicAdd segment-sum/scatter.

typedef __attribute__((ext_vector_type(16))) _Float16 v16h;
typedef __attribute__((ext_vector_type(8)))  float    v8f;

__device__ __forceinline__ unsigned enc_f32(float f) {
  unsigned u = __float_as_uint(f);
  return (u & 0x80000000u) ? ~u : (u | 0x80000000u);
}
__device__ __forceinline__ float dec_f32(unsigned u) {
  u = (u & 0x80000000u) ? (u & 0x7FFFFFFFu) : ~u;
  return __uint_as_float(u);
}

// ---------------------------------------------------------------- init scratch
__global__ void gt_init_kernel(unsigned* __restrict__ m1, float* __restrict__ s1,
                               unsigned* __restrict__ m2, float* __restrict__ s2, int N) {
  int i = blockIdx.x * blockDim.x + threadIdx.x;
  if (i < N * 4) { m1[i] = 0u; s1[i] = 0.0f; }   // enc(0) decodes to NaN -> "empty" sentinel
  if (i < N)     { m2[i] = 0u; s2[i] = 0.0f; }
}

// ------------------------------------------------- layer-1 fused q/k/v/skip GEMM
// grid.x = M tiles (16 rows), grid.y = which weight (0=q,1=k,2=v,3=skip)
// block = 256 threads = 8 waves; wave w computes output columns [16w,16w+16).
__global__ __launch_bounds__(256) void gt_gemm1_kernel(
    const float* __restrict__ x,
    const float* __restrict__ Wq, const float* __restrict__ bq,
    const float* __restrict__ Wk, const float* __restrict__ bk,
    const float* __restrict__ Wv, const float* __restrict__ bv,
    const float* __restrict__ Ws, const float* __restrict__ bs,
    float* __restrict__ oq, float* __restrict__ ok,
    float* __restrict__ ov, float* __restrict__ oo, int N) {
  __shared__ float xs[16 * 128];
  const int m0 = blockIdx.x * 16;
  const float* W; const float* b; float* o;
  switch (blockIdx.y) {
    case 0:  W = Wq; b = bq; o = oq; break;
    case 1:  W = Wk; b = bk; o = ok; break;
    case 2:  W = Wv; b = bv; o = ov; break;
    default: W = Ws; b = bs; o = oo; break;
  }
  for (int i = threadIdx.x; i < 16 * 128; i += 256) {
    int r = i >> 7, c = i & 127;
    int gr = m0 + r;
    xs[i] = (gr < N) ? x[(size_t)gr * 128 + c] : 0.0f;
  }
  __syncthreads();

  const int wave = threadIdx.x >> 5;
  const int lane = threadIdx.x & 31;
  const int hsel = lane >> 4;        // 0: low half, 1: high half
  const int l15  = lane & 15;
  const int col  = wave * 16 + l15;  // output column

  v8f acc = {};
#pragma unroll
  for (int kk = 0; kk < 128; kk += 32) {
    v16h afrag, bfrag;
    // A: 16-bit 16x32 layout; lanes 0-15: K {0..7,16..23}, lanes 16-31: +8
#pragma unroll
    for (int v = 0; v < 8; ++v) {
#pragma unroll
      for (int j = 0; j < 2; ++j) {
        int klocal = (v < 4) ? (2 * v + j) : (16 + 2 * (v - 4) + j);
        int k = kk + klocal + hsel * 8;
        afrag[2 * v + j] = (_Float16)xs[l15 * 128 + k];
      }
    }
    // B: 32x16, same K order as A; lanes 0-15: K {0..7,16..23}, lanes 16-31: +8
#pragma unroll
    for (int h = 0; h < 16; ++h) {
      int k = kk + ((h < 8) ? (hsel * 8 + h) : (16 + hsel * 8 + (h - 8)));
      bfrag[h] = (_Float16)W[(size_t)k * 128 + col];
    }
    acc = __builtin_amdgcn_wmma_f32_16x16x32_f16(false, afrag, false, bfrag,
                                                 (short)0, acc, false, false);
  }
  const float bias = b[col];
#pragma unroll
  for (int r = 0; r < 8; ++r) {   // D: VGPR r -> row r (lanes 0-15) / r+8 (lanes 16-31)
    int row = m0 + r + hsel * 8;
    if (row < N) o[(size_t)row * 128 + col] = acc[r] + bias;
  }
}

// --------------------------------------------- layer-1 edge logits + segment max
// one wave per edge; lane handles 4 channels; 8-lane groups = one head
__global__ __launch_bounds__(256) void gt_logits1_kernel(
    const float* __restrict__ q, const float* __restrict__ k,
    const int* __restrict__ src, const int* __restrict__ dst,
    float* __restrict__ alpha, unsigned* __restrict__ m, int E) {
  int e = blockIdx.x * (blockDim.x >> 5) + (threadIdx.x >> 5);
  if (e >= E) return;
  int lane = threadIdx.x & 31;
  int s = src[e], d = dst[e];
  const float4 qv = *(const float4*)(q + (size_t)d * 128 + lane * 4);
  const float4 kv = *(const float4*)(k + (size_t)s * 128 + lane * 4);
  float p = qv.x * kv.x + qv.y * kv.y + qv.z * kv.z + qv.w * kv.w;
  p += __shfl_xor(p, 1, 32);
  p += __shfl_xor(p, 2, 32);
  p += __shfl_xor(p, 4, 32);
  if ((lane & 7) == 0) {
    int h = lane >> 3;
    float av = p * 0.17677669529663687f;   // 1/sqrt(32)
    alpha[(size_t)e * 4 + h] = av;
    atomicMax(&m[(size_t)d * 4 + h], enc_f32(av));
  }
}

// ------------------------------------------------ layer-1 exp + segment sum
__global__ void gt_exp1_kernel(float* __restrict__ alpha, const unsigned* __restrict__ m,
                               float* __restrict__ ssum, const int* __restrict__ dst, int E) {
  int i = blockIdx.x * blockDim.x + threadIdx.x;
  if (i >= E * 4) return;
  int e = i >> 2, h = i & 3;
  int d = dst[e];
  float mv = dec_f32(m[(size_t)d * 4 + h]);
  if (!(mv > -INFINITY && mv < INFINITY)) mv = 0.0f;   // empty-segment guard (matches ref)
  float w = expf(alpha[i] - mv);
  alpha[i] = w;
  atomicAdd(&ssum[(size_t)d * 4 + h], w);
}

// ---------------------------------------- layer-1 weighted scatter into skip buf
__global__ __launch_bounds__(256) void gt_scatter1_kernel(
    const float* __restrict__ v, const float* __restrict__ alpha,
    const float* __restrict__ ssum, const int* __restrict__ src,
    const int* __restrict__ dst, float* __restrict__ out, int E) {
  int e = blockIdx.x * (blockDim.x >> 5) + (threadIdx.x >> 5);
  if (e >= E) return;
  int lane = threadIdx.x & 31;
  int s = src[e], d = dst[e];
  int h = lane >> 3;
  float a = alpha[(size_t)e * 4 + h] / (ssum[(size_t)d * 4 + h] + 1e-16f);
  const float4 vv = *(const float4*)(v + (size_t)s * 128 + lane * 4);
  float* op = out + (size_t)d * 128 + lane * 4;
  atomicAdd(op + 0, vv.x * a);
  atomicAdd(op + 1, vv.y * a);
  atomicAdd(op + 2, vv.z * a);
  atomicAdd(op + 3, vv.w * a);
}

// ------------------- layer-2: relu(h) @ {Wq2,Wk2,Wv2,Ws2} (128 -> 2 each), warp/node
__global__ __launch_bounds__(256) void gt_layer2_prep_kernel(
    const float* __restrict__ h1o,
    const float* __restrict__ Wq, const float* __restrict__ bq,
    const float* __restrict__ Wk, const float* __restrict__ bk,
    const float* __restrict__ Wv, const float* __restrict__ bv,
    const float* __restrict__ Ws, const float* __restrict__ bs,
    float* __restrict__ q2, float* __restrict__ k2, float* __restrict__ v2,
    float* __restrict__ out, int N) {
  int n = blockIdx.x * (blockDim.x >> 5) + (threadIdx.x >> 5);
  if (n >= N) return;
  int lane = threadIdx.x & 31;
  float4 hv = *(const float4*)(h1o + (size_t)n * 128 + lane * 4);
  float hcs[4] = {fmaxf(hv.x, 0.f), fmaxf(hv.y, 0.f), fmaxf(hv.z, 0.f), fmaxf(hv.w, 0.f)};
  const float* Wm[4] = {Wq, Wk, Wv, Ws};
  float acc[8] = {0, 0, 0, 0, 0, 0, 0, 0};
  int c0 = lane * 4;
#pragma unroll
  for (int j = 0; j < 4; ++j) {
    int c = c0 + j;
    float hc = hcs[j];
#pragma unroll
    for (int wmi = 0; wmi < 4; ++wmi) {
      acc[wmi * 2 + 0] += hc * Wm[wmi][c * 2 + 0];
      acc[wmi * 2 + 1] += hc * Wm[wmi][c * 2 + 1];
    }
  }
#pragma unroll
  for (int j = 0; j < 8; ++j) {
    acc[j] += __shfl_xor(acc[j], 16, 32);
    acc[j] += __shfl_xor(acc[j], 8, 32);
    acc[j] += __shfl_xor(acc[j], 4, 32);
    acc[j] += __shfl_xor(acc[j], 2, 32);
    acc[j] += __shfl_xor(acc[j], 1, 32);
  }
  if (lane == 0) {
    q2[(size_t)n * 2 + 0]  = acc[0] + bq[0]; q2[(size_t)n * 2 + 1]  = acc[1] + bq[1];
    k2[(size_t)n * 2 + 0]  = acc[2] + bk[0]; k2[(size_t)n * 2 + 1]  = acc[3] + bk[1];
    v2[(size_t)n * 2 + 0]  = acc[4] + bv[0]; v2[(size_t)n * 2 + 1]  = acc[5] + bv[1];
    out[(size_t)n * 2 + 0] = acc[6] + bs[0]; out[(size_t)n * 2 + 1] = acc[7] + bs[1];
  }
}

// ------------------------------------------------ layer-2 edge softmax + scatter
__global__ void gt_logits2_kernel(const float* __restrict__ q2, const float* __restrict__ k2,
                                  const int* __restrict__ src, const int* __restrict__ dst,
                                  float* __restrict__ alpha2, unsigned* __restrict__ m2, int E) {
  int e = blockIdx.x * blockDim.x + threadIdx.x;
  if (e >= E) return;
  int s = src[e], d = dst[e];
  float a = (q2[(size_t)d * 2] * k2[(size_t)s * 2] +
             q2[(size_t)d * 2 + 1] * k2[(size_t)s * 2 + 1]) * 0.7071067811865475f;
  alpha2[e] = a;
  atomicMax(&m2[d], enc_f32(a));
}

__global__ void gt_exp2_kernel(float* __restrict__ alpha2, const unsigned* __restrict__ m2,
                               float* __restrict__ s2, const int* __restrict__ dst, int E) {
  int e = blockIdx.x * blockDim.x + threadIdx.x;
  if (e >= E) return;
  int d = dst[e];
  float mv = dec_f32(m2[d]);
  if (!(mv > -INFINITY && mv < INFINITY)) mv = 0.0f;
  float w = expf(alpha2[e] - mv);
  alpha2[e] = w;
  atomicAdd(&s2[d], w);
}

__global__ void gt_scatter2_kernel(const float* __restrict__ v2, const float* __restrict__ alpha2,
                                   const float* __restrict__ s2, const int* __restrict__ src,
                                   const int* __restrict__ dst, float* __restrict__ out, int E) {
  int e = blockIdx.x * blockDim.x + threadIdx.x;
  if (e >= E) return;
  int s = src[e], d = dst[e];
  float a = alpha2[e] / (s2[d] + 1e-16f);
  atomicAdd(&out[(size_t)d * 2 + 0], v2[(size_t)s * 2 + 0] * a);
  atomicAdd(&out[(size_t)d * 2 + 1], v2[(size_t)s * 2 + 1] * a);
}

// ---------------------------------------------------------------------- launch
extern "C" void kernel_launch(void* const* d_in, const int* in_sizes, int n_in,
                              void* d_out, int out_size, void* d_ws, size_t ws_size,
                              hipStream_t stream) {
  const float* x   = (const float*)d_in[0];
  const float* Wq1 = (const float*)d_in[1];  const float* bq1 = (const float*)d_in[2];
  const float* Wk1 = (const float*)d_in[3];  const float* bk1 = (const float*)d_in[4];
  const float* Wv1 = (const float*)d_in[5];  const float* bv1 = (const float*)d_in[6];
  const float* Ws1 = (const float*)d_in[7];  const float* bs1 = (const float*)d_in[8];
  const float* Wq2 = (const float*)d_in[9];  const float* bq2 = (const float*)d_in[10];
  const float* Wk2 = (const float*)d_in[11]; const float* bk2 = (const float*)d_in[12];
  const float* Wv2 = (const float*)d_in[13]; const float* bv2 = (const float*)d_in[14];
  const float* Ws2 = (const float*)d_in[15]; const float* bs2 = (const float*)d_in[16];
  const int*   ei  = (const int*)d_in[17];

  const int N = in_sizes[0] / 128;
  const int E = in_sizes[17] / 2;
  const int* src = ei;
  const int* dst = ei + E;

  // workspace carve-up (all float-sized slots)
  float* ws = (float*)d_ws;
  size_t NC = (size_t)N * 128;
  float*    h1q    = ws;
  float*    h1k    = h1q + NC;
  float*    h1v    = h1k + NC;
  float*    h1o    = h1v + NC;                       // skip + aggregation (pre-relu h)
  float*    alpha1 = h1o + NC;                       // E*4
  unsigned* m1     = (unsigned*)(alpha1 + (size_t)E * 4);  // N*4
  float*    s1     = (float*)(m1 + (size_t)N * 4);         // N*4
  float*    q2     = s1 + (size_t)N * 4;             // N*2
  float*    k2     = q2 + (size_t)N * 2;
  float*    v2     = k2 + (size_t)N * 2;
  float*    alpha2 = v2 + (size_t)N * 2;             // E
  unsigned* m2     = (unsigned*)(alpha2 + (size_t)E);
  float*    s2     = (float*)(m2 + (size_t)N);
  float*    out    = (float*)d_out;

  const int T = 256;
  const int WPB = 8;  // waves per 256-thread block

  gt_init_kernel<<<(N * 4 + T - 1) / T, T, 0, stream>>>(m1, s1, m2, s2, N);

  dim3 g1((N + 15) / 16, 4);
  gt_gemm1_kernel<<<g1, 256, 0, stream>>>(x, Wq1, bq1, Wk1, bk1, Wv1, bv1, Ws1, bs1,
                                          h1q, h1k, h1v, h1o, N);

  gt_logits1_kernel<<<(E + WPB - 1) / WPB, 256, 0, stream>>>(h1q, h1k, src, dst, alpha1, m1, E);
  gt_exp1_kernel<<<(E * 4 + T - 1) / T, T, 0, stream>>>(alpha1, m1, s1, dst, E);
  gt_scatter1_kernel<<<(E + WPB - 1) / WPB, 256, 0, stream>>>(h1v, alpha1, s1, src, dst, h1o, E);

  gt_layer2_prep_kernel<<<(N + WPB - 1) / WPB, 256, 0, stream>>>(
      h1o, Wq2, bq2, Wk2, bk2, Wv2, bv2, Ws2, bs2, q2, k2, v2, out, N);

  gt_logits2_kernel<<<(E + T - 1) / T, T, 0, stream>>>(q2, k2, src, dst, alpha2, m2, E);
  gt_exp2_kernel<<<(E + T - 1) / T, T, 0, stream>>>(alpha2, m2, s2, dst, E);
  gt_scatter2_kernel<<<(E + T - 1) / T, T, 0, stream>>>(v2, alpha2, s2, src, dst, out, E);

  (void)n_in; (void)out_size; (void)ws_size;
}
